// GCNModel_89300960018653
// MI455X (gfx1250) — matrix operand order repaired and checked by reading, not compiled, not run on hardware
//
#include <hip/hip_runtime.h>
#include <hip/hip_bf16.h>

typedef __attribute__((ext_vector_type(16))) _Float16 v16h;
typedef __attribute__((ext_vector_type(8)))  _Float16 v8h;
typedef __attribute__((ext_vector_type(8)))  float    v8f;

#define NN 50000
#define NE 600000
#define HID 128
#define LDSTRIDE 136   // 128 + 8 halfs pad: row stride 272B -> lanes hit distinct banks

// ---------------- utility ----------------
__global__ void zero_kernel(float* __restrict__ p, long n) {
  long i = (long)blockIdx.x * blockDim.x + threadIdx.x;
  long st = (long)gridDim.x * blockDim.x;
  for (; i < n; i += st) p[i] = 0.0f;
}

// ---------------- gcn_norm ----------------
__global__ void deg_kernel(const int* __restrict__ col, const float* __restrict__ ew,
                           float* deg, int E) {
  int e = blockIdx.x * blockDim.x + threadIdx.x;
  if (e < E) atomicAdd(&deg[col[e]], ew[e]);
}

__global__ void dinv_kernel(float* deg, int N) {
  int i = blockIdx.x * blockDim.x + threadIdx.x;
  if (i < N) {
    float d = deg[i] + 1.0f;               // + self-loop weight 1.0
    deg[i] = (d > 0.0f) ? rsqrtf(d) : 0.0f;
  }
}

__global__ void norm_kernel(const int* __restrict__ row, const int* __restrict__ col,
                            const float* __restrict__ ew, const float* __restrict__ dinv,
                            float* __restrict__ norm, int E) {
  int e = blockIdx.x * blockDim.x + threadIdx.x;
  if (e < E) norm[e] = dinv[row[e]] * ew[e] * dinv[col[e]];
}

// ---------------- pack W (fp32 -> f16 WMMA B-fragment layout) ----------------
// B fragment 32x16 (KxN), this code's contiguous k order (same as A): lanes 0-15 hold K=0..15 (elem e -> k=e),
// lanes 16-31 hold K=16..31 (elem e -> k=16+e); column n = lane & 15.
// wp index: (((ntile*4 + kchunk)*32 + lane)*16 + e)  => contiguous 32B per lane.
__global__ void pack_w_kernel(const float* __restrict__ W, _Float16* __restrict__ wp) {
  int idx = blockIdx.x * blockDim.x + threadIdx.x;
  if (idx >= 8 * 4 * 32 * 16) return;
  int e    = idx & 15;
  int lane = (idx >> 4) & 31;
  int kc   = (idx >> 9) & 3;
  int t    = idx >> 11;
  int k = kc * 32 + ((lane >> 4) << 4) + e;
  int n = t * 16 + (lane & 15);
  wp[idx] = (_Float16)W[k * HID + n];
}

// ---------------- H = X @ W via WMMA ----------------
// Block = 256 threads (8 waves) -> one 16x128 output slab; wave w owns n-tile w.
// The 16x128 fp32 X tile is staged ONCE per block into LDS as f16 (4.25 KB with
// pad), so the 8 waves read A fragments via ds_load_b128 instead of 8x-redundant
// global traffic. Grid = 3125 exactly: no divergence -> EXEC all-ones for WMMA.
__global__ __launch_bounds__(256) void gemm_wmma_kernel(const float* __restrict__ X,
                                                        const _Float16* __restrict__ wp,
                                                        float* __restrict__ H) {
  __shared__ _Float16 smem[16 * LDSTRIDE];
  const int t = threadIdx.x;
  {
    // cooperative stage: thread t loads 8 consecutive floats of row (t>>4)
    int r  = t >> 4;
    int c8 = (t & 15) << 3;
    const float* xp = X + (long)(blockIdx.x * 16 + r) * HID + c8;
    float4 x0 = *(const float4*)(xp);
    float4 x1 = *(const float4*)(xp + 4);
    v8h h8;
    h8[0]=(_Float16)x0.x; h8[1]=(_Float16)x0.y; h8[2]=(_Float16)x0.z; h8[3]=(_Float16)x0.w;
    h8[4]=(_Float16)x1.x; h8[5]=(_Float16)x1.y; h8[6]=(_Float16)x1.z; h8[7]=(_Float16)x1.w;
    *(v8h*)(&smem[r * LDSTRIDE + c8]) = h8;   // 16B-aligned (272 = 17*16)
  }
  __syncthreads();

  const int lane = t & 31;
  const int wave = t >> 5;                    // n-tile 0..7
  const int mloc = lane & 15;
  const int ka   = (lane >> 4) << 3;          // C/D row offset: lanes 16-31 hold M+8
  v8f c = {};
#pragma unroll
  for (int kc = 0; kc < 4; ++kc) {
    // A fragment (16x32 f16): e=0..15 -> k = base+16*(lane>=16)+e
    const _Float16* ap = &smem[mloc * LDSTRIDE + kc * 32 + ((lane >> 4) << 4)];
    v8h lo = *(const v8h*)(ap);
    v8h hi = *(const v8h*)(ap + 8);
    v16h a = __builtin_shufflevector(lo, hi, 0,1,2,3,4,5,6,7,8,9,10,11,12,13,14,15);
    v16h b = *(const v16h*)(wp + (((wave * 4 + kc) * 32 + lane) << 4));
    c = __builtin_amdgcn_wmma_f32_16x16x32_f16(false, a, false, b, (short)0, c, false, false);
  }
  // C/D layout: VGPR j -> M = j + 8*(lane>=16), N = lane&15
  const int n  = wave * 16 + mloc;
  const long mr = (long)blockIdx.x * 16 + ka;
#pragma unroll
  for (int j = 0; j < 8; ++j) H[(mr + j) * HID + n] = c[j];
}

// ---------------- scatter: agg[col] += norm * h[row] ----------------
// One wave per edge. Edge index is wave-uniform: force it scalar so row/col/norm
// become SMEM loads instead of 96 redundant per-lane VMEM loads per wave.
__global__ void scatter_kernel(const int* __restrict__ row, const int* __restrict__ col,
                               const float* __restrict__ norm, const float* __restrict__ H,
                               float* agg) {
  long t = (long)blockIdx.x * blockDim.x + threadIdx.x;
  int e = (int)(t >> 5);
  if (e >= NE) return;
  e = __builtin_amdgcn_readfirstlane(e);
  int f = ((int)(t & 31)) << 2;
  int r = row[e], cdst = col[e];
  float nv = norm[e];
  float4 h4 = *(const float4*)(H + (long)r * HID + f);
  float* ap = agg + (long)cdst * HID + f;
  atomicAdd(ap + 0, h4.x * nv);
  atomicAdd(ap + 1, h4.y * nv);
  atomicAdd(ap + 2, h4.z * nv);
  atomicAdd(ap + 3, h4.w * nv);
}

// ---------------- finalize: x = relu(agg + dinv^2*h + b [+ xres]) ----------------
__global__ void finalize_kernel(const float* __restrict__ agg, const float* __restrict__ H,
                                const float* __restrict__ dinv, const float* __restrict__ bias,
                                const float* __restrict__ xres, float* __restrict__ xout,
                                int residual) {
  long t = (long)blockIdx.x * blockDim.x + threadIdx.x;
  int i = (int)(t >> 5);
  if (i >= NN) return;
  i = __builtin_amdgcn_readfirstlane(i);       // wave-uniform node index
  int f = ((int)(t & 31)) << 2;
  float di = dinv[i];
  float sl = di * di;                          // self-loop norm
  float4 a4 = *(const float4*)(agg + (long)i * HID + f);
  float4 h4 = *(const float4*)(H   + (long)i * HID + f);
  float4 b4 = *(const float4*)(bias + f);
  float4 v;
  v.x = fmaf(sl, h4.x, a4.x) + b4.x;
  v.y = fmaf(sl, h4.y, a4.y) + b4.y;
  v.z = fmaf(sl, h4.z, a4.z) + b4.z;
  v.w = fmaf(sl, h4.w, a4.w) + b4.w;
  if (residual) {
    float4 r4 = *(const float4*)(xres + (long)i * HID + f);
    v.x += r4.x; v.y += r4.y; v.z += r4.z; v.w += r4.w;
  }
  v.x = fmaxf(v.x, 0.0f); v.y = fmaxf(v.y, 0.0f);
  v.z = fmaxf(v.z, 0.0f); v.w = fmaxf(v.w, 0.0f);
  *(float4*)(xout + (long)i * HID + f) = v;
}

// ---------------- mean pool + FC (algebraically reordered) ----------------
__global__ void colsum_kernel(const float* __restrict__ x, float* cs) {
  int f  = threadIdx.x & 127;
  int i0 = blockIdx.x * 2 + (threadIdx.x >> 7);
  float s = 0.0f;
  for (int i = i0; i < NN; i += gridDim.x * 2) s += x[(long)i * HID + f];
  atomicAdd(&cs[f], s);
}

__global__ void final_kernel(const float* __restrict__ cs, const float* __restrict__ Wfc,
                             const float* __restrict__ bfc, float* __restrict__ out) {
  int j = threadIdx.x;
  if (j < 2) {
    float s = 0.0f;
    for (int f = 0; f < HID; ++f) s += cs[f] * Wfc[f * 2 + j];
    out[j] = s / (float)NN + bfc[j];
  }
}

// ---------------- launch ----------------
extern "C" void kernel_launch(void* const* d_in, const int* in_sizes, int n_in,
                              void* d_out, int out_size, void* d_ws, size_t ws_size,
                              hipStream_t stream) {
  const float* node = (const float*)d_in[0];
  const int*   edges= (const int*)  d_in[1];   // [2, NE] flat
  const float* ea   = (const float*)d_in[2];
  const float* W1   = (const float*)d_in[3];
  const float* b1   = (const float*)d_in[4];
  const float* Ws   = (const float*)d_in[5];   // [4,128,128]
  const float* bs   = (const float*)d_in[6];   // [4,128]
  const float* Wfc  = (const float*)d_in[7];   // [128,2]
  const float* bfc  = (const float*)d_in[8];   // [2]

  const int* row = edges;        // edges[0] = source
  const int* col = edges + NE;   // edges[1] = target

  char* base = (char*)d_ws;
  size_t off = 0;
  auto alloc = [&](size_t bytes) { size_t r = off; off += (bytes + 255) & ~(size_t)255; return r; };
  float*    dinv = (float*)   (base + alloc((size_t)NN * 4));
  float*    norm = (float*)   (base + alloc((size_t)NE * 4));
  float*    H    = (float*)   (base + alloc((size_t)NN * HID * 4));
  float*    AGG  = (float*)   (base + alloc((size_t)NN * HID * 4));
  float*    XA   = (float*)   (base + alloc((size_t)NN * HID * 4));
  _Float16* WP   = (_Float16*)(base + alloc((size_t)8 * 4 * 32 * 16 * 2));
  float*    CS   = (float*)   (base + alloc(128 * 4));

  // gcn_norm
  zero_kernel<<<196, 256, 0, stream>>>(dinv, (long)NN);
  deg_kernel<<<(NE + 255) / 256, 256, 0, stream>>>(col, ea, dinv, NE);
  dinv_kernel<<<(NN + 255) / 256, 256, 0, stream>>>(dinv, NN);
  norm_kernel<<<(NE + 255) / 256, 256, 0, stream>>>(row, col, ea, dinv, norm, NE);

  auto conv = [&](const float* Xin, const float* Wf, const float* bias, int residual) {
    pack_w_kernel<<<64, 256, 0, stream>>>(Wf, WP);
    gemm_wmma_kernel<<<NN / 16, 256, 0, stream>>>(Xin, WP, H);
    zero_kernel<<<4096, 256, 0, stream>>>(AGG, (long)NN * HID);
    scatter_kernel<<<(NE * 32) / 256, 256, 0, stream>>>(row, col, norm, H, AGG);
    finalize_kernel<<<(NN * 32) / 256, 256, 0, stream>>>(AGG, H, dinv, bias, Xin, XA, residual);
  };

  conv(node, W1, b1, /*residual=*/0);
  for (int s = 0; s < 4; ++s)
    conv(XA, Ws + (size_t)s * HID * HID, bs + (size_t)s * HID, /*residual=*/1);

  zero_kernel<<<1, 128, 0, stream>>>(CS, 128);
  colsum_kernel<<<256, 256, 0, stream>>>(XA, CS);
  final_kernel<<<1, 32, 0, stream>>>(CS, Wfc, bfc, (float*)d_out);
}
